// MeshTextureNet_43748536877377
// MI455X (gfx1250) — compile-verified, matrix operand order checked by reading, not run on hardware
//
#include <hip/hip_runtime.h>

// ============================================================================
// MeshTextureNet forward for gfx1250 (MI455X, wave32, WMMA).
// GEMMs: v_wmma_f32_16x16x32_bf16, f32 accumulate. Weights pre-packed to
// zero-padded bf16 [Mpad][Kpad]; activations K-padded (and column-padded for
// the conv5 layer) so the staging loops are guard-free straight-line code.
// LDS double-buffered (80B pitch = bank-conflict-free fragment reads), one
// barrier per K-step, global prefetch overlapped with WMMA.
// ============================================================================

typedef __attribute__((ext_vector_type(16))) __bf16 v16bf;
typedef __attribute__((ext_vector_type(8)))  float  v8f;

#define RSQ_BN 0.99999500003749969f   // 1/sqrt(1 + 1e-5)

__device__ __forceinline__ unsigned short f2bf(float f) {
    unsigned int u = __float_as_uint(f);
    unsigned int r = u + 0x7FFFu + ((u >> 16) & 1u);   // round-to-nearest-even
    return (unsigned short)(r >> 16);
}

// ---------------------------------------------------------------------------
// Fused GEMM + bias + BN + ReLU.  Wb: bf16 [Mpad][Kpad] zero-padded weights.
//   Y[z][m][n] = relu((sum_k W[m][k]*Xsel(z,k,n) + bias[m]) * g[m]*RSQ + bb[m])
// KMODE 0: Xsel = X[z*xbs + k*ldx + n]       (rows k are zero for k >= C)
// KMODE 1: conv1d k=5 pad=2 on column-padded X: idx = ci*ldx + n + t - 2
// ---------------------------------------------------------------------------
template <int BM, int BN, int KMODE>
__global__ __launch_bounds__(256) void gemm_bn_t(
    const unsigned short* __restrict__ Wb, const float* __restrict__ X,
    const float* __restrict__ bias, const float* __restrict__ g,
    const float* __restrict__ bb, float* __restrict__ Y,
    int M, int Kpad, int N, int ldx, int ldy, int xbs, int ybs)
{
    constexpr int WMr = BM / 4, WNr = BN / 2;
    constexpr int FM  = WMr / 16, FN = WNr / 16;
    constexpr int AQ  = (BM * 32) / (256 * 8);  // uint4 quads per thread (A)
    constexpr int BE  = (BN * 32) / 256;        // elements per thread (B)

    __shared__ unsigned short As[2][BM * 40];   // ping-pong, pitch 40 shorts
    __shared__ unsigned short Bs[2][BN * 40];

    const int tid  = threadIdx.x;
    const int wave = tid >> 5, lane = tid & 31;
    const int half = lane >> 4, l16 = lane & 15;
    const int wm = (wave >> 1) * WMr, wn = (wave & 1) * WNr;
    const int nBase = blockIdx.x * BN, mBase = blockIdx.y * BM;

    const float* Xz = X + (size_t)blockIdx.z * (size_t)xbs;
    float*       Yz = Y + (size_t)blockIdx.z * (size_t)ybs;

    // precomputed per-thread staging addresses (guard-free by construction)
    int aIdx[AQ], aOff[AQ];
#pragma unroll
    for (int j = 0; j < AQ; ++j) {
        int i = tid + j * 256, row = i >> 2, q = i & 3;
        aIdx[j] = (mBase + row) * Kpad + q * 8;
        aOff[j] = row * 40 + q * 8;
    }
    int bIdx[BE], bOff[BE];
#pragma unroll
    for (int j = 0; j < BE; ++j) {
        int i = tid + j * 256, kk = i / BN, nn = i % BN;
        bIdx[j] = kk * ldx + nBase + nn - (KMODE ? 2 : 0);
        bOff[j] = nn * 40 + kk;
    }

    v8f acc[FM][FN];
#pragma unroll
    for (int fi = 0; fi < FM; ++fi)
#pragma unroll
        for (int fj = 0; fj < FN; ++fj)
            acc[fi][fj] = v8f{0.f, 0.f, 0.f, 0.f, 0.f, 0.f, 0.f, 0.f};

    const int nT = Kpad >> 5;

    // prologue: tile 0 -> LDS[0]
    {
        uint4 wa[AQ]; float vb[BE];
#pragma unroll
        for (int j = 0; j < AQ; ++j) wa[j] = *(const uint4*)(Wb + aIdx[j]);
#pragma unroll
        for (int j = 0; j < BE; ++j) vb[j] = Xz[bIdx[j]];
#pragma unroll
        for (int j = 0; j < AQ; ++j) *(uint4*)&As[0][aOff[j]] = wa[j];
#pragma unroll
        for (int j = 0; j < BE; ++j) Bs[0][bOff[j]] = f2bf(vb[j]);
    }
    __syncthreads();

    for (int it = 0; it < nT; ++it) {
        const int p = it & 1;
        uint4 wa[AQ]; float vb[BE];
        const bool more = (it + 1 < nT);
        if (more) {                               // prefetch next tile (overlaps wmma)
            int kb = (it + 1) << 5;
            int boff = KMODE ? ((kb & 1023) * ldx + (kb >> 10)) : kb * ldx;
#pragma unroll
            for (int j = 0; j < AQ; ++j) wa[j] = *(const uint4*)(Wb + aIdx[j] + kb);
#pragma unroll
            for (int j = 0; j < BE; ++j) vb[j] = Xz[bIdx[j] + boff];
        }

        // fragments: lane(m=l16, half) K ranges [8h,8h+8) U [16+8h, +8)
        v16bf a[FM], b[FN];
#pragma unroll
        for (int fi = 0; fi < FM; ++fi) {
            const unsigned short* ap = &As[p][(wm + fi * 16 + l16) * 40];
            ((uint4*)&a[fi])[0] = *(const uint4*)(ap + half * 8);
            ((uint4*)&a[fi])[1] = *(const uint4*)(ap + 16 + half * 8);
        }
#pragma unroll
        for (int fj = 0; fj < FN; ++fj) {
            const unsigned short* bp = &Bs[p][(wn + fj * 16 + l16) * 40];
            ((uint4*)&b[fj])[0] = *(const uint4*)(bp + half * 8);
            ((uint4*)&b[fj])[1] = *(const uint4*)(bp + 16 + half * 8);
        }
#pragma unroll
        for (int fi = 0; fi < FM; ++fi)
#pragma unroll
            for (int fj = 0; fj < FN; ++fj)
                acc[fi][fj] = __builtin_amdgcn_wmma_f32_16x16x32_bf16(
                    false, a[fi], false, b[fj], (short)0, acc[fi][fj], false, false);

        if (more) {                               // store prefetch to other buffer
            const int q = p ^ 1;
#pragma unroll
            for (int j = 0; j < AQ; ++j) *(uint4*)&As[q][aOff[j]] = wa[j];
#pragma unroll
            for (int j = 0; j < BE; ++j) Bs[q][bOff[j]] = f2bf(vb[j]);
        }
        __syncthreads();
    }

    // epilogue: VGPR r -> M = r + 8*half, N = lane%16
#pragma unroll
    for (int fi = 0; fi < FM; ++fi) {
#pragma unroll
        for (int r = 0; r < 8; ++r) {
            int m = mBase + wm + fi * 16 + r + 8 * half;
            if (m < M) {
                float bi = bias[m], gm = g[m] * RSQ_BN, bm = bb[m];
#pragma unroll
                for (int fj = 0; fj < FN; ++fj) {
                    int n = nBase + wn + fj * 16 + l16;
                    float v = acc[fi][fj][r] + bi;
                    Yz[m * ldy + n] = fmaxf(0.f, v * gm + bm);
                }
            }
        }
    }
}

// --- weight repack: f32 [M][K] (or conv5 [M][1024][5]) -> bf16 [Mpad][Kpad] -
__global__ void wrepack_kernel(const float* __restrict__ src,
                               unsigned short* __restrict__ dst,
                               int M, int K, int Kpad, int total, int conv5)
{
    int i = blockIdx.x * 256 + threadIdx.x;
    if (i >= total) return;
    int m = i / Kpad, k = i % Kpad;
    float v = 0.f;
    if (m < M && k < K)
        v = conv5 ? src[(m * 1024 + (k & 1023)) * 5 + (k >> 10)] : src[m * K + k];
    dst[i] = f2bf(v);
}

// --- zero fill / column pad ------------------------------------------------
__global__ void zerof_kernel(float* __restrict__ p, int total) {
    int i = blockIdx.x * 256 + threadIdx.x;
    if (i < total) p[i] = 0.f;
}
__global__ void zcols_kernel(float* __restrict__ p, int rows, int ld) {
    int i = blockIdx.x * 256 + threadIdx.x;
    if (i >= rows * 4) return;
    int r = i >> 2, c = i & 3;
    int col = (c < 2) ? c : (ld - 4 + (c & 1));      // 0,1,ld-4,ld-3
    p[(size_t)r * ld + col] = 0.f;
}

// ---------------------------------------------------------------------------
// tc1: 7x7 conv (3->16), pad 3, + BN + ReLU on 512x512.
// ---------------------------------------------------------------------------
__global__ __launch_bounds__(256) void tc1_kernel(
    const float* __restrict__ img, const float* __restrict__ w,
    const float* __restrict__ bias, const float* __restrict__ g,
    const float* __restrict__ bb, float* __restrict__ out)
{
    const int H = 512, W = 512;
    __shared__ float tile[3][22][22];
    __shared__ float wsm[16 * 3 * 49];
    const int x0 = blockIdx.x * 16 - 3, y0 = blockIdx.y * 16 - 3, b = blockIdx.z;

    for (int i = threadIdx.x; i < 3 * 22 * 22; i += 256) {
        int c = i / 484, rem = i % 484, yy = rem / 22, xx = rem % 22;
        int gy = y0 + yy, gx = x0 + xx;
        tile[c][yy][xx] = (gx >= 0 && gx < W && gy >= 0 && gy < H)
            ? img[((size_t)(b * 3 + c) * H + gy) * W + gx] : 0.f;
    }
    for (int i = threadIdx.x; i < 16 * 3 * 49; i += 256) wsm[i] = w[i];
    __syncthreads();

    const int tx = threadIdx.x & 15, ty = threadIdx.x >> 4;
    for (int co = 0; co < 16; ++co) {
        float acc = bias[co];
        for (int c = 0; c < 3; ++c)
            for (int ky = 0; ky < 7; ++ky)
#pragma unroll
                for (int kx = 0; kx < 7; ++kx)
                    acc += tile[c][ty + ky][tx + kx] * wsm[(co * 3 + c) * 49 + ky * 7 + kx];
        out[((size_t)(b * 16 + co) * H + (blockIdx.y * 16 + ty)) * W + (blockIdx.x * 16 + tx)]
            = fmaxf(0.f, acc * (g[co] * RSQ_BN) + bb[co]);
    }
}

// ---------------------------------------------------------------------------
// tc2 (3x3 16->16, pad1, BN+ReLU) fused with tc3 (1x1 16->16, BN+ReLU).
// ---------------------------------------------------------------------------
__global__ __launch_bounds__(256) void tc23_kernel(
    const float* __restrict__ in,
    const float* __restrict__ w2, const float* __restrict__ b2,
    const float* __restrict__ g2, const float* __restrict__ bb2,
    const float* __restrict__ w3, const float* __restrict__ b3,
    const float* __restrict__ g3, const float* __restrict__ bb3,
    float* __restrict__ out)
{
    const int H = 512, W = 512;
    __shared__ float tile[16][18][18];
    __shared__ float wsm[16 * 16 * 9];
    __shared__ float w3sm[256];
    const int x0 = blockIdx.x * 16 - 1, y0 = blockIdx.y * 16 - 1, b = blockIdx.z;

    for (int i = threadIdx.x; i < 16 * 18 * 18; i += 256) {
        int c = i / 324, rem = i % 324, yy = rem / 18, xx = rem % 18;
        int gy = y0 + yy, gx = x0 + xx;
        tile[c][yy][xx] = (gx >= 0 && gx < W && gy >= 0 && gy < H)
            ? in[((size_t)(b * 16 + c) * H + gy) * W + gx] : 0.f;
    }
    for (int i = threadIdx.x; i < 2304; i += 256) wsm[i] = w2[i];
    if (threadIdx.x < 256) w3sm[threadIdx.x] = w3[threadIdx.x];
    __syncthreads();

    const int tx = threadIdx.x & 15, ty = threadIdx.x >> 4;
    float mid[16];
    for (int co = 0; co < 16; ++co) {
        float acc = b2[co];
        for (int c = 0; c < 16; ++c) {
            const float* wp = &wsm[(co * 16 + c) * 9];
#pragma unroll
            for (int ky = 0; ky < 3; ++ky)
#pragma unroll
                for (int kx = 0; kx < 3; ++kx)
                    acc += tile[c][ty + ky][tx + kx] * wp[ky * 3 + kx];
        }
        mid[co] = fmaxf(0.f, acc * (g2[co] * RSQ_BN) + bb2[co]);
    }
    for (int o = 0; o < 16; ++o) {
        float acc = b3[o];
#pragma unroll
        for (int c = 0; c < 16; ++c) acc += mid[c] * w3sm[o * 16 + c];
        out[((size_t)(b * 16 + o) * H + (blockIdx.y * 16 + ty)) * W + (blockIdx.x * 16 + tx)]
            = fmaxf(0.f, acc * (g3[o] * RSQ_BN) + bb3[o]);
    }
}

// ---------------------------------------------------------------------------
// Fused per-face texture tail (grid_sample -> tx1 -> tx2+avgpool -> tx3+max).
// ---------------------------------------------------------------------------
__global__ __launch_bounds__(128) void face_tex_kernel(
    const float* __restrict__ t3, const float* __restrict__ uv,
    const float* __restrict__ w1, const float* __restrict__ b1,
    const float* __restrict__ g1, const float* __restrict__ bb1,
    const float* __restrict__ w2, const float* __restrict__ b2,
    const float* __restrict__ g2, const float* __restrict__ bb2,
    const float* __restrict__ w3, const float* __restrict__ b3,
    const float* __restrict__ g3, const float* __restrict__ bb3,
    float* __restrict__ tex, int Nn)
{
    const int H = 512, W = 512;
    const int face = blockIdx.x;
    const int b = face / Nn, n = face % Nn;
    __shared__ float samp[16][49];
    __shared__ float x1[32][49];
    __shared__ float pool[32][9];

    for (int i = threadIdx.x; i < 16 * 49; i += 128) {
        int c = i / 49, pt = i % 49;
        const float* gp = &uv[((size_t)face * 49 + pt) * 3];
        float ix = (gp[0] + 1.f) * 0.5f * (W - 1);
        float iy = (gp[1] + 1.f) * 0.5f * (H - 1);
        float ix0 = floorf(ix), iy0 = floorf(iy);
        float wx1 = ix - ix0, wy1 = iy - iy0;
        const float* img = t3 + (size_t)(b * 16 + c) * H * W;
        float acc = 0.f;
#pragma unroll
        for (int dy = 0; dy < 2; ++dy)
#pragma unroll
            for (int dx = 0; dx < 2; ++dx) {
                float fy = iy0 + dy, fx = ix0 + dx;
                float wt = (dx ? wx1 : 1.f - wx1) * (dy ? wy1 : 1.f - wy1);
                if (fx >= 0.f && fx <= (float)(W - 1) && fy >= 0.f && fy <= (float)(H - 1)) {
                    int xi = (int)fminf(fmaxf(fx, 0.f), (float)(W - 1));
                    int yi = (int)fminf(fmaxf(fy, 0.f), (float)(H - 1));
                    acc += wt * img[yi * W + xi];
                }
            }
        samp[c][pt] = acc;
    }
    __syncthreads();

    for (int i = threadIdx.x; i < 32 * 49; i += 128) {
        int co = i / 49, pt = i % 49, py = pt / 7, px = pt % 7;
        float acc = b1[co];
        for (int c = 0; c < 16; ++c) {
            const float* wp = &w1[(co * 16 + c) * 9];
#pragma unroll
            for (int ky = 0; ky < 3; ++ky) {
                int sy = py + ky - 1; if (sy < 0 || sy > 6) continue;
#pragma unroll
                for (int kx = 0; kx < 3; ++kx) {
                    int sx = px + kx - 1; if (sx < 0 || sx > 6) continue;
                    acc += samp[c][sy * 7 + sx] * wp[ky * 3 + kx];
                }
            }
        }
        x1[co][pt] = fmaxf(0.f, acc * (g1[co] * RSQ_BN) + bb1[co]);
    }
    __syncthreads();

    // tx2 1x1 on zero-padded 9x9 (pad positions still get bias) + avgpool/9
    for (int i = threadIdx.x; i < 32 * 9; i += 128) {
        int co = i / 9, p = i % 9, oy = p / 3, ox = p % 3;
        float s = 0.f;
        for (int yy = 3 * oy - 1; yy <= 3 * oy + 1; ++yy) {
            if (yy < 0 || yy > 8) continue;
            for (int xx = 3 * ox - 1; xx <= 3 * ox + 1; ++xx) {
                if (xx < 0 || xx > 8) continue;
                float acc = b2[co];
                int sy = yy - 1, sx = xx - 1;
                if (sy >= 0 && sy < 7 && sx >= 0 && sx < 7) {
                    const float* wp = &w2[co * 32];
                    for (int c = 0; c < 32; ++c) acc += x1[c][sy * 7 + sx] * wp[c];
                }
                s += fmaxf(0.f, acc * (g2[co] * RSQ_BN) + bb2[co]);
            }
        }
        pool[co][p] = s * (1.f / 9.f);
    }
    __syncthreads();

    for (int o = threadIdx.x; o < 64; o += 128) {
        float mx = -INFINITY;
        const float* wp = &w3[o * 32];
        for (int p = 0; p < 9; ++p) {
            float acc = b3[o];
            for (int c = 0; c < 32; ++c) acc += pool[c][p] * wp[c];
            mx = fmaxf(mx, fmaxf(0.f, acc * (g3[o] * RSQ_BN) + bb3[o]));
        }
        tex[(b * 64 + o) * Nn + n] = mx;
    }
}

// ---------------------------------------------------------------------------
// curve_desc: gather ring normals, l2norm, project, max, BN+ReLU.
// ---------------------------------------------------------------------------
__global__ __launch_bounds__(64) void curve_kernel(
    const float* __restrict__ normals, const int* __restrict__ ring,
    const float* __restrict__ dirs, const float* __restrict__ g,
    const float* __restrict__ bb, float* __restrict__ out, int kk, int Nn)
{
    const int b = blockIdx.y, n = blockIdx.x, tid = threadIdx.x;
    __shared__ float allv[13][3];
    if (tid <= kk) {
        int src = (tid < kk) ? ring[(b * Nn + n) * kk + tid] : n;
        float x = normals[(b * 3 + 0) * Nn + src];
        float y = normals[(b * 3 + 1) * Nn + src];
        float z = normals[(b * 3 + 2) * Nn + src];
        float r = fmaxf(sqrtf(x * x + y * y + z * z), 1e-12f);
        allv[tid][0] = x / r; allv[tid][1] = y / r; allv[tid][2] = z / r;
    }
    __syncthreads();
    const int m = tid;
    float d0 = dirs[m], d1 = dirs[64 + m], d2 = dirs[128 + m];
    float dn = fmaxf(sqrtf(d0 * d0 + d1 * d1 + d2 * d2), 1e-12f);
    d0 /= dn; d1 /= dn; d2 /= dn;
    float mx = -INFINITY;
    for (int j = 0; j <= kk; ++j)
        mx = fmaxf(mx, allv[j][0] * d0 + allv[j][1] * d1 + allv[j][2] * d2);
    out[(b * 64 + m) * Nn + n] = fmaxf(0.f, mx * (g[m] * RSQ_BN) + bb[m]);
}

// --- fpr helpers -----------------------------------------------------------
__global__ void fpr_prep_kernel(const float* __restrict__ corners,
                                float* __restrict__ cbuf, int Nn)
{
    int i = blockIdx.x * 256 + threadIdx.x;
    const int NB3 = 3 * 4 * Nn;
    if (i >= 6 * NB3) return;
    int row = i / NB3, col = i % NB3;
    int v = col / (4 * Nn), rem = col % (4 * Nn);
    int b = rem / Nn, n = rem % Nn;
    int srcRow = (v == 0) ? row : (v == 1) ? (row + 3)
               : ((row < 3) ? (row + 6) : (row - 3));
    cbuf[row * NB3 + col] = corners[(b * 9 + srcRow) * Nn + n];
}

__global__ void avg3_kernel(const float* __restrict__ r2o,
                            float* __restrict__ fea, int Nn)
{
    int i = blockIdx.x * 256 + threadIdx.x;
    const int NB = 4 * Nn;
    if (i >= 32 * NB) return;
    int c = i / NB, rem = i % NB, b = rem / Nn, n = rem % Nn;
    const float* rr = r2o + c * 3 * NB + b * Nn + n;
    fea[(b * 32 + c) * Nn + n] = (rr[0] + rr[NB] + rr[2 * NB]) * (1.f / 3.f);
}

// --- channel concat copy (dstC = padded batch stride in rows) --------------
__global__ void copyc_kernel(const float* __restrict__ src, float* __restrict__ dst,
                             int srcC, int dstC, int off, int Nn)
{
    int i = blockIdx.x * 256 + threadIdx.x;
    if (i >= 4 * srcC * Nn) return;
    int b = i / (srcC * Nn), rem = i % (srcC * Nn);
    int c = rem / Nn, n = rem % Nn;
    dst[((size_t)b * dstC + off + c) * Nn + n] = src[i];
}

// --- mesh_conv neighbor stats: dst rows [st | s_sum | dif | div], stride Cp -
__global__ void meshcat_kernel(const float* __restrict__ st, const int* __restrict__ nidx,
                               float* __restrict__ dst, int C, int Cp, int Nn)
{
    int i = blockIdx.x * 256 + threadIdx.x;
    if (i >= 4 * C * Nn) return;
    int b = i / (C * Nn), rem = i % (C * Nn);
    int c = rem / Nn, n = rem % Nn;
    const int* idx = &nidx[(b * Nn + n) * 3];
    const float* stb = st + (b * C + c) * Nn;
    float v0 = stb[idx[0]], v1 = stb[idx[1]], v2 = stb[idx[2]], s = stb[n];
    float ssum = v0 + v1 + v2;
    float dif  = fabsf(v2 - v1) + fabsf(v1 - v0) + fabsf(v0 - v1); // per reference
    float dv   = fabsf(v0 - s) + fabsf(v1 - s) + fabsf(v2 - s);
    size_t base = ((size_t)b * Cp + c) * Nn + n;
    size_t cs = (size_t)C * Nn;
    dst[base] = s; dst[base + cs] = ssum; dst[base + 2 * cs] = dif; dst[base + 3 * cs] = dv;
}

// --- o2: 1x256 dot per column ----------------------------------------------
__global__ void o2_kernel(const float* __restrict__ y, const float* __restrict__ w,
                          const float* __restrict__ bsc, float* __restrict__ out, int Nn)
{
    int i = blockIdx.x * 256 + threadIdx.x;
    if (i >= 4 * Nn) return;
    int b = i / Nn, n = i % Nn;
    const float* yb = y + b * 256 * Nn + n;
    float acc = bsc[0];
    for (int c = 0; c < 256; ++c) acc += w[c] * yb[c * Nn];
    out[i] = acc;
}

// ============================================================================
// Host orchestration
// ============================================================================
extern "C" void kernel_launch(void* const* d_in, const int* in_sizes, int n_in,
                              void* d_out, int out_size, void* d_ws, size_t ws_size,
                              hipStream_t stream)
{
    (void)n_in; (void)out_size; (void)ws_size;
    const int B = 4;
    const int Nn = in_sizes[0] / (B * 3);          // 8192
    const int NB = B * Nn;

    const float* centers = (const float*)d_in[0];
    const float* normals = (const float*)d_in[1];
    const float* corners = (const float*)d_in[2];
    const int*   ring1   = (const int*)d_in[5];
    const int*   ring2   = (const int*)d_in[6];
    const int*   ring3   = (const int*)d_in[7];
    const int*   nidx    = (const int*)d_in[8];
    const float* texture = (const float*)d_in[9];
    const float* uvgrid  = (const float*)d_in[10];
    auto P = [&](int i) { return (const float*)d_in[i]; };
    enum { D1 = 11, D2 = 14, D3 = 17, R1 = 20, R2 = 24, F1 = 28, F2 = 32,
           S1 = 36, S2 = 40, CF1 = 44, CF2 = 48, MC1C = 52, MC1K = 56, MC1A = 60,
           MC2C = 64, MC2K = 68, MC2A = 72, FM = 76, CM = 80, TC1 = 84, TC2 = 88,
           TC3 = 92, TX1 = 96, TX2 = 100, TX3 = 104, O1W = 108, O2W = 112 };

    size_t off = 0;
    auto allocB = [&](size_t bytes) -> void* {
        void* p = (char*)d_ws + off;
        off = (off + bytes + 255) & ~(size_t)255;
        return p;
    };
    auto alloc = [&](size_t elems) -> float* { return (float*)allocB(elems * 4); };

    // ---- pre-pack all weights to zero-padded bf16 [Mpad][Kpad] ----
    struct WBuf { const unsigned short* p; int M, Kpad; };
    auto prep = [&](int idx, int M, int K, int Kpad, int conv5) -> WBuf {
        int Mpad = ((M + 127) / 128) * 128;
        unsigned short* dst = (unsigned short*)allocB((size_t)Mpad * Kpad * 2);
        int total = Mpad * Kpad;
        wrepack_kernel<<<(total + 255) / 256, 256, 0, stream>>>(P(idx), dst, M, K, Kpad, total, conv5);
        return WBuf{dst, M, Kpad};
    };
    WBuf wS1 = prep(S1, 64, 3, 32, 0),    wS2 = prep(S2, 64, 64, 64, 0);
    WBuf wR1 = prep(R1, 32, 6, 32, 0),    wR2 = prep(R2, 32, 32, 32, 0);
    WBuf wF1 = prep(F1, 64, 32, 32, 0),   wF2 = prep(F2, 64, 64, 64, 0);
    WBuf wCF1 = prep(CF1, 131, 256, 256, 0), wCF2 = prep(CF2, 131, 131, 160, 0);
    WBuf wM1C = prep(MC1C, 256, 259, 288, 0), wM1K = prep(MC1K, 131, 524, 544, 0);
    WBuf wM1A = prep(MC1A, 256, 131, 160, 0);
    WBuf wM2C = prep(MC2C, 512, 576, 576, 0), wM2K = prep(MC2K, 256, 1024, 1024, 0);
    WBuf wM2A = prep(MC2A, 512, 256, 256, 0);
    WBuf wFM = prep(FM, 1024, 1024, 1024, 0), wCM = prep(CM, 1024, 1792, 1792, 0);
    WBuf wO1 = prep(O1W, 256, 5120, 5120, 1);

    // ---- persistent activation buffers ----
    float* tex  = alloc((size_t)B * 64 * Nn);
    float* sp0  = alloc((size_t)B * 64 * Nn);
    float* st0  = alloc((size_t)B * 131 * Nn);
    float* sp1  = alloc((size_t)B * 256 * Nn);
    float* st1  = alloc((size_t)B * 256 * Nn);
    float* sp2  = alloc((size_t)B * 512 * Nn);
    float* st2  = alloc((size_t)B * 512 * Nn);
    float* fmo  = alloc((size_t)B * 1024 * Nn);
    float* cmoR = alloc((size_t)B * 1024 * (Nn + 4));   // column-padded
    float* cmo  = cmoR + 2;
    float* o1o  = alloc((size_t)B * 256 * Nn);
    const size_t scratch = off;

    auto gemm = [&](WBuf w, const float* X, int pw, float* Y, int Nc, int Bz,
                    int kmode, int ldx, int ldy, int xbs, int ybs) {
        if (w.M >= 128) {
            dim3 grid(Nc / 64, (w.M + 127) / 128, Bz);
            if (kmode)
                gemm_bn_t<128, 64, 1><<<grid, 256, 0, stream>>>(
                    w.p, X, P(pw + 1), P(pw + 2), P(pw + 3), Y, w.M, w.Kpad, Nc, ldx, ldy, xbs, ybs);
            else
                gemm_bn_t<128, 64, 0><<<grid, 256, 0, stream>>>(
                    w.p, X, P(pw + 1), P(pw + 2), P(pw + 3), Y, w.M, w.Kpad, Nc, ldx, ldy, xbs, ybs);
        } else {
            dim3 grid(Nc / 32, (w.M + 63) / 64, Bz);
            gemm_bn_t<64, 32, 0><<<grid, 256, 0, stream>>>(
                w.p, X, P(pw + 1), P(pw + 2), P(pw + 3), Y, w.M, w.Kpad, Nc, ldx, ldy, xbs, ybs);
        }
    };
    // common case: ldx = ldy = Nc
    auto gemmP = [&](WBuf w, const float* X, int pw, float* Y, int Nc, int Bz,
                     int xbs, int ybs) {
        gemm(w, X, pw, Y, Nc, Bz, 0, Nc, Nc, xbs, ybs);
    };
    auto copyc = [&](const float* src, float* dst, int srcC, int dstC, int o) {
        int tot = B * srcC * Nn;
        copyc_kernel<<<(tot + 255) / 256, 256, 0, stream>>>(src, dst, srcC, dstC, o, Nn);
    };
    auto zerof = [&](float* p, size_t cnt) {
        zerof_kernel<<<(int)((cnt + 255) / 256), 256, 0, stream>>>(p, (int)cnt);
    };
    auto zpad = [&](float* buf, int C, int Cp) {
        for (int b = 0; b < B; ++b)
            zerof(buf + ((size_t)b * Cp + C) * Nn, (size_t)(Cp - C) * Nn);
    };

    // ---- Phase T: texture pyramid + per-face fused texture tail ----
    {
        off = scratch;
        float* t1 = alloc((size_t)B * 16 * 512 * 512);
        float* t2 = alloc((size_t)B * 16 * 512 * 512);
        dim3 g2d(32, 32, B);
        tc1_kernel<<<g2d, 256, 0, stream>>>(texture, P(TC1), P(TC1 + 1), P(TC1 + 2), P(TC1 + 3), t1);
        tc23_kernel<<<g2d, 256, 0, stream>>>(t1, P(TC2), P(TC2 + 1), P(TC2 + 2), P(TC2 + 3),
                                             P(TC3), P(TC3 + 1), P(TC3 + 2), P(TC3 + 3), t2);
        face_tex_kernel<<<B * Nn, 128, 0, stream>>>(t2, uvgrid,
            P(TX1), P(TX1 + 1), P(TX1 + 2), P(TX1 + 3),
            P(TX2), P(TX2 + 1), P(TX2 + 2), P(TX2 + 3),
            P(TX3), P(TX3 + 1), P(TX3 + 2), P(TX3 + 3), tex, Nn);
    }

    // ---- Phase A: sp0, fpr, curves, cf1/cf2 -> st0 ----
    {
        off = scratch;
        float* ctrP  = alloc((size_t)B * 32 * Nn);      // centers K-padded to 32
        float* a64   = alloc((size_t)B * 64 * Nn);
        float* cbuf  = alloc((size_t)32 * 3 * NB);      // K-padded to 32
        float* r1o   = alloc((size_t)32 * 3 * NB);
        float* r2o   = alloc((size_t)32 * 3 * NB);
        float* fea   = alloc((size_t)B * 32 * Nn);
        float* f1o   = alloc((size_t)B * 64 * Nn);
        float* st0a  = alloc((size_t)B * 64 * Nn);
        float* c1    = alloc((size_t)B * 64 * Nn);
        float* c2    = alloc((size_t)B * 64 * Nn);
        float* c3    = alloc((size_t)B * 64 * Nn);
        float* cat256 = alloc((size_t)B * 256 * Nn);
        float* cf1o  = alloc((size_t)B * 160 * Nn);     // K-padded to 160

        zpad(ctrP, 3, 32);
        copyc(centers, ctrP, 3, 32, 0);
        gemmP(wS1, ctrP, S1, a64, Nn, B, 32 * Nn, 64 * Nn);
        gemmP(wS2, a64, S2, sp0, Nn, B, 64 * Nn, 64 * Nn);

        int tot = 6 * 3 * NB;
        fpr_prep_kernel<<<(tot + 255) / 256, 256, 0, stream>>>(corners, cbuf, Nn);
        zerof(cbuf + (size_t)6 * 3 * NB, (size_t)26 * 3 * NB);
        gemmP(wR1, cbuf, R1, r1o, 3 * NB, 1, 0, 0);
        gemmP(wR2, r1o, R2, r2o, 3 * NB, 1, 0, 0);
        tot = 32 * NB;
        avg3_kernel<<<(tot + 255) / 256, 256, 0, stream>>>(r2o, fea, Nn);
        gemmP(wF1, fea, F1, f1o, Nn, B, 32 * Nn, 64 * Nn);
        gemmP(wF2, f1o, F2, st0a, Nn, B, 64 * Nn, 64 * Nn);

        dim3 cg(Nn, B);
        curve_kernel<<<cg, 64, 0, stream>>>(normals, ring1, P(D1), P(D1 + 1), P(D1 + 2), c1, 3, Nn);
        curve_kernel<<<cg, 64, 0, stream>>>(normals, ring2, P(D2), P(D2 + 1), P(D2 + 2), c2, 6, Nn);
        curve_kernel<<<cg, 64, 0, stream>>>(normals, ring3, P(D3), P(D3 + 1), P(D3 + 2), c3, 12, Nn);

        copyc(st0a, cat256, 64, 256, 0);
        copyc(c1, cat256, 64, 256, 64);
        copyc(c2, cat256, 64, 256, 128);
        copyc(c3, cat256, 64, 256, 192);
        zpad(cf1o, 131, 160);
        gemmP(wCF1, cat256, CF1, cf1o, Nn, B, 256 * Nn, 160 * Nn);
        gemmP(wCF2, cf1o, CF2, st0, Nn, B, 160 * Nn, 131 * Nn);
    }

    // ---- Phase M1: mesh_conv 1 ----
    {
        off = scratch;
        float* comb = alloc((size_t)B * 288 * Nn);      // 259 -> 288
        float* cat  = alloc((size_t)B * 544 * Nn);      // 524 -> 544
        float* cato = alloc((size_t)B * 160 * Nn);      // 131 -> 160
        zpad(comb, 259, 288);
        copyc(sp0, comb, 64, 288, 0);
        copyc(st0, comb, 131, 288, 64);
        copyc(tex, comb, 64, 288, 195);
        gemmP(wM1C, comb, MC1C, sp1, Nn, B, 288 * Nn, 256 * Nn);
        zpad(cat, 524, 544);
        int tot = B * 131 * Nn;
        meshcat_kernel<<<(tot + 255) / 256, 256, 0, stream>>>(st0, nidx, cat, 131, 544, Nn);
        zpad(cato, 131, 160);
        gemmP(wM1K, cat, MC1K, cato, Nn, B, 544 * Nn, 160 * Nn);
        gemmP(wM1A, cato, MC1A, st1, Nn, B, 160 * Nn, 256 * Nn);
    }

    // ---- Phase M2: mesh_conv 2 ----
    {
        off = scratch;
        float* comb = alloc((size_t)B * 576 * Nn);
        float* cat  = alloc((size_t)B * 1024 * Nn);
        float* cato = alloc((size_t)B * 256 * Nn);
        copyc(sp1, comb, 256, 576, 0);
        copyc(st1, comb, 256, 576, 256);
        copyc(tex, comb, 64, 576, 512);
        gemmP(wM2C, comb, MC2C, sp2, Nn, B, 576 * Nn, 512 * Nn);
        int tot = B * 256 * Nn;
        meshcat_kernel<<<(tot + 255) / 256, 256, 0, stream>>>(st1, nidx, cat, 256, 1024, Nn);
        gemmP(wM2K, cat, MC2K, cato, Nn, B, 1024 * Nn, 256 * Nn);
        gemmP(wM2A, cato, MC2A, st2, Nn, B, 256 * Nn, 512 * Nn);
    }

    // ---- Phase F: fm, cm (column-padded out), o1 (conv5), o2 ----
    {
        off = scratch;
        float* fmin = alloc((size_t)B * 1024 * Nn);
        copyc(sp2, fmin, 512, 1024, 0);
        copyc(st2, fmin, 512, 1024, 512);
        gemmP(wFM, fmin, FM, fmo, Nn, B, 1024 * Nn, 1024 * Nn);
    }
    {
        off = scratch;
        float* cmin = alloc((size_t)B * 1792 * Nn);
        copyc(fmo, cmin, 1024, 1792, 0);
        copyc(sp1, cmin, 256, 1792, 1024);
        copyc(sp2, cmin, 512, 1792, 1280);
        int rows = B * 1024;
        zcols_kernel<<<(rows * 4 + 255) / 256, 256, 0, stream>>>(cmoR, rows, Nn + 4);
        gemm(wCM, cmin, CM, cmo, Nn, B, 0, /*ldx=*/Nn, /*ldy=*/Nn + 4,
             /*xbs=*/1792 * Nn, /*ybs=*/1024 * (Nn + 4));
        // o1: conv5 on column-padded cmo, fully guard-free B loader
        gemm(wO1, cmo, O1W, o1o, Nn, B, 1, /*ldx=*/Nn + 4, /*ldy=*/Nn,
             /*xbs=*/1024 * (Nn + 4), /*ybs=*/256 * Nn);
        int tot = B * Nn;
        o2_kernel<<<(tot + 255) / 256, 256, 0, stream>>>(o1o, P(O2W), P(O2W + 1),
                                                         (float*)d_out, Nn);
    }
}
